// DecoderNetwork_39719857553640
// MI455X (gfx1250) — compile-verified, matrix operand order checked by reading, not run on hardware
//
#include <hip/hip_runtime.h>
#include <hip/hip_bf16.h>
#include <math.h>

// ---------------------------------------------------------------------------
// DecoderNetwork on MI455X (gfx1250, wave32, WMMA).
// Memory-bound (~6.5 TFLOP vs ~100us of HBM traffic at 23.3 TB/s), so:
//  - one generic batched WMMA-bf16 GEMM with fused epilogues
//  - vectorized (b128) global staging, packed f32->bf16 convert, double-
//    buffered LDS so HBM latency hides behind ds_load+v_wmma work
//  - decoder attention restructured: o2 = att @ (dWp @ src2^T)
// All GEMM dims here are multiples of the 128x128 tile / 32 K-step, so the
// staging path carries no bounds checks (this is what keeps it coalesced
// b128 loads instead of exec-masked scalar loads).
// Scratch requirement: 6*128MB + 640MB ~= 1.47 GB in d_ws.
// ---------------------------------------------------------------------------

#define Bdim 8
#define Tdim 1024
#define Ddim 4096
#define Hdim 4
#define HSdim 1024
#define BT   (Bdim * Tdim)     // 8192
#define FFN  (5 * Ddim)        // 20480

typedef __bf16 bf16_t;
typedef __attribute__((ext_vector_type(16))) __bf16       v16bf;
typedef __attribute__((ext_vector_type(4)))  __bf16       v4bf;
typedef __attribute__((ext_vector_type(8)))  float        v8f;
typedef __attribute__((ext_vector_type(4)))  unsigned int u32x4;

#define BM 128
#define BN 128
#define BK 32
#define KPAD 40   // LDS row stride (bf16): 32 + 8 pad; 80B rows, 16B-aligned

__device__ __forceinline__ v4bf cvt4(float4 v)
{
    v4bf r;
    r[0] = (bf16_t)v.x; r[1] = (bf16_t)v.y;
    r[2] = (bf16_t)v.z; r[3] = (bf16_t)v.w;
    return r;
}

// ---- tiled batched GEMM:  C = epi(A @ opB(B) + bias + res) ----------------
// A: [M,K] fp32 row-major (lda). B: opB=0 -> [K,N] (ldb), opB=1 -> [N,K] (ldb).
// Block tile 128x128, K-step 32 (== WMMA K for bf16). 8 wave32s; each wave
// computes a 64x32 sub-tile as 4x2 fragments of v_wmma_f32_16x16x32_bf16.
// M, N must be multiples of 128 and K a multiple of 32 (true for all calls).
__global__ __launch_bounds__(256)
void gemm_wmma_bf16(const float* __restrict__ A, const float* __restrict__ Bm,
                    float* __restrict__ C,
                    const float* __restrict__ bias, const float* __restrict__ res,
                    int M, int N, int K, int lda, int ldb, int ldc,
                    int opB, int epiGelu, int batchI,
                    long long sAo, long long sAi, long long sBo, long long sBi,
                    long long sCo, long long sCi,
                    long long sBiasO, long long sBiasI)
{
    __shared__ __align__(16) bf16_t As[2][BM][KPAD];
    __shared__ __align__(16) bf16_t Bs[2][BN][KPAD];

    const int z  = blockIdx.z;
    const int oz = z / batchI;
    const int iz = z - oz * batchI;
    A  += oz * sAo + iz * sAi;
    Bm += oz * sBo + iz * sBi;
    C  += oz * sCo + iz * sCi;
    if (bias) bias += oz * sBiasO + iz * sBiasI;
    if (res)  res  += oz * sCo    + iz * sCi;     // residual mirrors C layout

    const int tid  = threadIdx.x;
    const int lane = tid & 31;
    const int wave = tid >> 5;
    const int wm   = wave >> 2;          // 0..1 -> 64-row slab
    const int wn   = wave & 3;           // 0..3 -> 32-col slab
    const int lane15   = lane & 15;
    const int laneHalf = lane >> 4;      // K-half select per ISA A/B layout
    const int blockM = blockIdx.y * BM;
    const int blockN = blockIdx.x * BN;

    // staging coordinates
    const int rA  = tid >> 3;            // 0..31   (row / n group, +j*32)
    const int kA  = (tid & 7) * 4;       // 0,4,..28
    const int kB0 = tid >> 5;            // 0..7    (k group for opB=0, +j*8)
    const int n4  = (tid & 31) * 4;      // 0..124  (n group for opB=0)

    float4 ra[4], rb[4];

    auto stage_load = [&](int kt) {
#pragma unroll
        for (int j = 0; j < 4; ++j)
            ra[j] = *(const float4*)&A[(size_t)(blockM + rA + j * 32) * lda + kt + kA];
        if (opB == 0) {
#pragma unroll
            for (int j = 0; j < 4; ++j)
                rb[j] = *(const float4*)&Bm[(size_t)(kt + kB0 + j * 8) * ldb + blockN + n4];
        } else {
#pragma unroll
            for (int j = 0; j < 4; ++j)
                rb[j] = *(const float4*)&Bm[(size_t)(blockN + rA + j * 32) * ldb + kt + kA];
        }
    };
    auto stage_store = [&](int buf) {
#pragma unroll
        for (int j = 0; j < 4; ++j)
            *(v4bf*)&As[buf][rA + j * 32][kA] = cvt4(ra[j]);
        if (opB == 0) {
#pragma unroll
            for (int j = 0; j < 4; ++j) {
                int kk = kB0 + j * 8;
                Bs[buf][n4 + 0][kk] = (bf16_t)rb[j].x;
                Bs[buf][n4 + 1][kk] = (bf16_t)rb[j].y;
                Bs[buf][n4 + 2][kk] = (bf16_t)rb[j].z;
                Bs[buf][n4 + 3][kk] = (bf16_t)rb[j].w;
            }
        } else {
#pragma unroll
            for (int j = 0; j < 4; ++j)
                *(v4bf*)&Bs[buf][rA + j * 32][kA] = cvt4(rb[j]);
        }
    };

    v8f acc[4][2];
#pragma unroll
    for (int tm = 0; tm < 4; ++tm)
#pragma unroll
        for (int tn = 0; tn < 2; ++tn)
            acc[tm][tn] = (v8f){};

    // prologue: stage tile 0
    stage_load(0);
    stage_store(0);
    __syncthreads();

    int cur = 0;
    for (int kt = 0; kt < K; kt += BK) {
        const bool more = (kt + BK) < K;
        if (more) stage_load(kt + BK);   // issue next-tile HBM loads early

        // ---- fragment loads from LDS: ISA 16-bit A/B layout ----
        // lane 0-15 : row/col lane15, K = {0..7, 16..23}
        // lane 16-31: row/col lane15, K = {8..15, 24..31}
        const int kb = laneHalf * 8;
        v16bf afrag[4], bfrag[2];
#pragma unroll
        for (int tm = 0; tm < 4; ++tm) {
            int r = wm * 64 + tm * 16 + lane15;
            union { u32x4 u[2]; v16bf v; } t;
            t.u[0] = *(const u32x4*)&As[cur][r][kb];
            t.u[1] = *(const u32x4*)&As[cur][r][kb + 16];
            afrag[tm] = t.v;
        }
#pragma unroll
        for (int tn = 0; tn < 2; ++tn) {
            int c = wn * 32 + tn * 16 + lane15;
            union { u32x4 u[2]; v16bf v; } t;
            t.u[0] = *(const u32x4*)&Bs[cur][c][kb];
            t.u[1] = *(const u32x4*)&Bs[cur][c][kb + 16];
            bfrag[tn] = t.v;
        }
#pragma unroll
        for (int tm = 0; tm < 4; ++tm)
#pragma unroll
            for (int tn = 0; tn < 2; ++tn)
                acc[tm][tn] = __builtin_amdgcn_wmma_f32_16x16x32_bf16(
                    false, afrag[tm], false, bfrag[tn],
                    (short)0, acc[tm][tn], false, false);

        if (more) {
            stage_store(cur ^ 1);        // safe: buf cur^1 last read before
            __syncthreads();             // the previous iteration's barrier
            cur ^= 1;
        }
    }

    // ---- epilogue + store (C/D layout: VGPR j -> row j + laneHalf*8) ----
#pragma unroll
    for (int tm = 0; tm < 4; ++tm) {
#pragma unroll
        for (int tn = 0; tn < 2; ++tn) {
            int col = blockN + wn * 32 + tn * 16 + lane15;
            if (col >= N) continue;
            float bcol = bias ? bias[col] : 0.0f;
#pragma unroll
            for (int j = 0; j < 8; ++j) {
                int row = blockM + wm * 64 + tm * 16 + laneHalf * 8 + j;
                if (row >= M) continue;
                float v = acc[tm][tn][j] + bcol;
                if (res) v += res[(size_t)row * ldc + col];
                if (epiGelu)
                    v = 0.5f * v * (1.0f + erff(v * 0.70710678118654752f));
                C[(size_t)row * ldc + col] = v;
            }
        }
    }
}

// ---- in-place causal softmax over last dim; scale folded in ---------------
__global__ __launch_bounds__(256)
void softmax_causal(float* __restrict__ s, int T, float scale)
{
    const int row = blockIdx.x;          // (h*B + b)*T + t
    const int t   = row % T;
    float* p = s + (size_t)row * T;
    const int tid = threadIdx.x;
    const int cnt = T >> 8;              // T/256 (<=8)
    float vals[8];

    __shared__ float red[256];
    float m = -INFINITY;
    for (int j = 0; j < cnt; ++j) {
        int idx = tid + (j << 8);
        float v = (idx <= t) ? p[idx] * scale : -INFINITY;
        vals[j] = v;
        m = fmaxf(m, v);
    }
    red[tid] = m; __syncthreads();
    for (int off = 128; off > 0; off >>= 1) {
        if (tid < off) red[tid] = fmaxf(red[tid], red[tid + off]);
        __syncthreads();
    }
    m = red[0]; __syncthreads();

    float sum = 0.0f;
    for (int j = 0; j < cnt; ++j) {
        float e = __expf(vals[j] - m);   // exp(-inf) == 0 for masked lanes
        vals[j] = e;
        sum += e;
    }
    red[tid] = sum; __syncthreads();
    for (int off = 128; off > 0; off >>= 1) {
        if (tid < off) red[tid] += red[tid + off];
        __syncthreads();
    }
    float inv = 1.0f / red[0];
    for (int j = 0; j < cnt; ++j)
        p[tid + (j << 8)] = vals[j] * inv;
}

// ---- LayerNorm: t = x (+ addend);  out = mode ? t + LN(t) : LN(t) ---------
__global__ __launch_bounds__(256)
void ln_kernel(const float* __restrict__ x, const float* __restrict__ addend,
               const float* __restrict__ g, const float* __restrict__ bta,
               float* __restrict__ out, int mode)
{
    const size_t row = blockIdx.x;
    const float* xr = x + row * Ddim;
    const float* ar = addend ? addend + row * Ddim : nullptr;
    const int tid = threadIdx.x;
    float t[16];
    float s = 0.0f, ss = 0.0f;
#pragma unroll
    for (int j = 0; j < 16; ++j) {
        int d = tid + (j << 8);
        float v = xr[d] + (ar ? ar[d] : 0.0f);
        t[j] = v; s += v; ss += v * v;
    }
    __shared__ float r0[256], r1[256];
    r0[tid] = s; r1[tid] = ss; __syncthreads();
    for (int off = 128; off > 0; off >>= 1) {
        if (tid < off) { r0[tid] += r0[tid + off]; r1[tid] += r1[tid + off]; }
        __syncthreads();
    }
    const float mean = r0[0] * (1.0f / Ddim);
    const float var  = r1[0] * (1.0f / Ddim) - mean * mean;
    const float rstd = rsqrtf(var + 1e-5f);
#pragma unroll
    for (int j = 0; j < 16; ++j) {
        int d = tid + (j << 8);
        float ln = (t[j] - mean) * rstd * g[d] + bta[d];
        out[row * Ddim + d] = mode ? (t[j] + ln) : ln;
    }
}

// ---- bias2[h,b,u] = dot(dbp[h,:], src2[b,u,:]) ----------------------------
__global__ __launch_bounds__(256)
void gemv_bias2(const float* __restrict__ dbp, const float* __restrict__ src2,
                float* __restrict__ out)
{
    const int z = blockIdx.x;                         // h*B*T + b*T + u
    const int h   = z / (Bdim * Tdim);
    const int rem = z - h * (Bdim * Tdim);
    const float* w = dbp + (size_t)h * Ddim;
    const float* xv = src2 + (size_t)rem * Ddim;
    const int tid = threadIdx.x;
    float s = 0.0f;
    for (int d = tid; d < Ddim; d += 256) s += w[d] * xv[d];
    __shared__ float red[256];
    red[tid] = s; __syncthreads();
    for (int off = 128; off > 0; off >>= 1) {
        if (tid < off) red[tid] += red[tid + off];
        __syncthreads();
    }
    if (tid == 0) out[z] = red[0];
}

// ---------------------------------------------------------------------------
static inline void launch_gemm(hipStream_t st,
    const float* A, const float* Bm, float* C,
    const float* bias, const float* res,
    int M, int N, int K, int lda, int ldb, int ldc,
    int opB, int gelu, int O, int I,
    long long sAo, long long sAi, long long sBo, long long sBi,
    long long sCo, long long sCi, long long sBiasO, long long sBiasI)
{
    dim3 grid((N + BN - 1) / BN, (M + BM - 1) / BM, O * I);
    gemm_wmma_bf16<<<grid, 256, 0, st>>>(A, Bm, C, bias, res,
        M, N, K, lda, ldb, ldc, opB, gelu, I,
        sAo, sAi, sBo, sBi, sCo, sCi, sBiasO, sBiasI);
}

extern "C" void kernel_launch(void* const* d_in, const int* in_sizes, int n_in,
                              void* d_out, int out_size, void* d_ws, size_t ws_size,
                              hipStream_t stream)
{
    (void)in_sizes; (void)n_in; (void)out_size; (void)ws_size;
    const float* src = (const float*)d_in[0];
    const float* att = (const float*)d_in[1];
    const float* mWq = (const float*)d_in[2];
    const float* mbq = (const float*)d_in[3];
    const float* mWk = (const float*)d_in[4];
    const float* mbk = (const float*)d_in[5];
    const float* mWv = (const float*)d_in[6];
    const float* dWp = (const float*)d_in[7];
    const float* dbp = (const float*)d_in[8];
    const float* dWo = (const float*)d_in[9];
    const float* dbo = (const float*)d_in[10];
    const float* fW1 = (const float*)d_in[11];
    const float* fb1 = (const float*)d_in[12];
    const float* fW2 = (const float*)d_in[13];
    const float* fb2 = (const float*)d_in[14];
    const float* g1  = (const float*)d_in[15];
    const float* b1  = (const float*)d_in[16];
    const float* g2  = (const float*)d_in[17];
    const float* b2  = (const float*)d_in[18];
    float* out = (float*)d_out;

    // scratch carving (floats). SZ = 33,554,432 floats = 128 MB.
    const size_t SZ = (size_t)Hdim * Bdim * Tdim * HSdim;
    float* ws     = (float*)d_ws;
    float* qb     = ws;            // [H,B,T,HS]   -> later Mbuf, later src_f
    float* kb     = ws + 1 * SZ;   // [H,B,T,HS]   -> later o2cat
    float* vb     = ws + 2 * SZ;   // [H,B,T,HS]   -> later bias2, later h_in
    float* scores = ws + 3 * SZ;   // [H,B,T,T]    -> later trg_f
    float* obuf   = ws + 4 * SZ;   // [B,T,D]      -> later trg
    float* src2   = ws + 5 * SZ;   // [B,T,D]
    float* hbuf   = ws + 6 * SZ;   // [B*T, 5D] (640 MB)
    float* Mbuf   = qb;
    float* o2cat  = kb;
    float* bias2  = vb;            // only H*B*T floats, freed before h_in
    float* h_in   = vb;
    float* trg    = obuf;
    float* trg_f  = scores;
    float* src_f  = qb;

    const long long llBT_HS = (long long)BT * HSdim;
    const long long llT_HS  = (long long)Tdim * HSdim;
    const long long llT_T   = (long long)Tdim * Tdim;
    const long long llT_D   = (long long)Tdim * Ddim;

    // 1) q/k/v = src @ mW*[h] (+bias)          M=8192 N=1024 K=4096, batch H
    launch_gemm(stream, src, mWq, qb, mbq, nullptr,
        BT, HSdim, Ddim, Ddim, HSdim, HSdim, 0, 0, Hdim, 1,
        0, 0, (long long)Ddim * HSdim, 0, llBT_HS, 0, HSdim, 0);
    launch_gemm(stream, src, mWk, kb, mbk, nullptr,
        BT, HSdim, Ddim, Ddim, HSdim, HSdim, 0, 0, Hdim, 1,
        0, 0, (long long)Ddim * HSdim, 0, llBT_HS, 0, HSdim, 0);
    launch_gemm(stream, src, mWv, vb, nullptr, nullptr,
        BT, HSdim, Ddim, Ddim, HSdim, HSdim, 0, 0, Hdim, 1,
        0, 0, (long long)Ddim * HSdim, 0, llBT_HS, 0, 0, 0);

    // 2) scores[h,b] = q @ k^T                 M=N=K=1024, batch (H,B)
    launch_gemm(stream, qb, kb, scores, nullptr, nullptr,
        Tdim, Tdim, HSdim, HSdim, HSdim, Tdim, 1, 0, Hdim, Bdim,
        llBT_HS, llT_HS, llBT_HS, llT_HS,
        (long long)Bdim * llT_T, llT_T, 0, 0);

    // 3) causal softmax * sqrt(HS), in place
    softmax_causal<<<Hdim * Bdim * Tdim, 256, 0, stream>>>(scores, Tdim, 32.0f);

    // 4) o[b,t,h*HS+e] = probs @ v             write heads interleaved into D
    launch_gemm(stream, scores, vb, obuf, nullptr, nullptr,
        Tdim, HSdim, Tdim, Tdim, HSdim, Ddim, 0, 0, Hdim, Bdim,
        (long long)Bdim * llT_T, llT_T, llBT_HS, llT_HS,
        HSdim, llT_D, 0, 0);

    // 5) src2 = (src+o) + LN(src+o, g1,b1)
    ln_kernel<<<BT, 256, 0, stream>>>(src, obuf, g1, b1, src2, 1);

    // 6) Mbuf[h,b] = dWp[h] @ src2[b]^T        M=1024 N=1024 K=4096
    launch_gemm(stream, dWp, src2, Mbuf, nullptr, nullptr,
        HSdim, Tdim, Ddim, Ddim, Ddim, Tdim, 1, 0, Hdim, Bdim,
        (long long)HSdim * Ddim, 0, 0, llT_D,
        (long long)Bdim * llT_T, llT_T, 0, 0);

    // 7) bias2[h,b,u] = dbp[h] . src2[b,u,:]
    gemv_bias2<<<Hdim * Bdim * Tdim, 256, 0, stream>>>(dbp, src2, bias2);

    // 8) o2cat[b,t,h*T+u] = att[b] @ Mbuf[h,b] + bias2[h,b,:]
    launch_gemm(stream, att, Mbuf, o2cat, bias2, nullptr,
        Tdim, Tdim, HSdim, HSdim, Tdim, Ddim, 0, 0, Hdim, Bdim,
        0, llT_HS, (long long)Bdim * llT_T, llT_T,
        Tdim, llT_D, (long long)Bdim * Tdim, Tdim);

    // 9) trg = src2 + (o2cat @ dWo + dbo)      M=8192 N=K=4096
    launch_gemm(stream, o2cat, dWo, trg, dbo, src2,
        BT, Ddim, Ddim, Ddim, Ddim, Ddim, 0, 0, 1, 1,
        0, 0, 0, 0, 0, 0, 0, 0);

    // 10) trg_f = trg + LN(trg, g1,b1)
    ln_kernel<<<BT, 256, 0, stream>>>(trg, nullptr, g1, b1, trg_f, 1);

    // 11) h_in = LN(trg_f, g2,b2)
    ln_kernel<<<BT, 256, 0, stream>>>(trg_f, nullptr, g2, b2, h_in, 0);

    // 12) hbuf = GELU(h_in @ fW1 + fb1)        M=8192 N=20480 K=4096
    launch_gemm(stream, h_in, fW1, hbuf, fb1, nullptr,
        BT, FFN, Ddim, Ddim, FFN, FFN, 0, 1, 1, 1,
        0, 0, 0, 0, 0, 0, 0, 0);

    // 13) src_f = trg_f + (hbuf @ fW2 + fb2)   M=8192 N=4096 K=20480
    launch_gemm(stream, hbuf, fW2, src_f, fb2, trg_f,
        BT, Ddim, FFN, FFN, Ddim, Ddim, 0, 0, 1, 1,
        0, 0, 0, 0, 0, 0, 0, 0);

    // 14) out = LN(src_f, g2,b2)
    ln_kernel<<<BT, 256, 0, stream>>>(src_f, nullptr, g2, b2, out, 0);
}
